// get_model_53927609369124
// MI455X (gfx1250) — matrix operand order repaired and checked by reading, not compiled, not run on hardware
//
#include <hip/hip_runtime.h>
#include <hip/hip_bf16.h>

// ---------------------------------------------------------------------------
// PointNet++ (get_model) forward for MI455X / gfx1250.
// Grouped inputs gathered once to compact f16; each SA stage's MLP chain runs
// fused in LDS with v_wmma_f32_16x16x32_f16; weights pre-swizzled into WMMA
// fragment order (2x global_load_b128 per B fragment); input tiles pulled
// into LDS with global_load_async_to_lds_b128 (ASYNCcnt).
// ---------------------------------------------------------------------------

typedef __attribute__((ext_vector_type(16))) _Float16 v16h;
typedef __attribute__((ext_vector_type(8)))  float    v8f;

#define THREADS 256
#define NWAVES  8

// ---------------------------------------------------------------------------
// WMMA fragment loaders (layouts per CDNA5 ISA §7.12.2, wave32)
// ---------------------------------------------------------------------------

// A: 16x32 f16 (MxK). lane l: M = l&15 ; K-halfword packing:
//   v<4 : K = 2v + 8*(l>>4) ; v>=4 : K = 16 + 2(v-4) + 8*(l>>4)
// -> two contiguous 16B runs per lane; compiler emits 2x ds_load_b128.
__device__ inline v16h load_a_frag(const _Float16* X, int ld, int m0, int k0, int lane) {
    v16h a;
    const _Float16* row = X + (m0 + (lane & 15)) * ld + k0 + ((lane >> 4) << 3);
#pragma unroll
    for (int v = 0; v < 8; ++v) {
        int kb = (v < 4) ? (2 * v) : (16 + 2 * (v - 4));
        a[2 * v]     = row[kb];
        a[2 * v + 1] = row[kb + 1];
    }
    return a;
}

// ---------------------------------------------------------------------------
// Descriptor for a fused MLP chain (up to 4 layers)
// ---------------------------------------------------------------------------
struct MlpDesc {
    const _Float16* W[4];     // fragment-packed f16 weights (see pack_w_frag)
    const float*    bias[4];
    const float*    gamma[4];
    const float*    beta[4];
    int             width[4];
    int             nlayers;
    int             cin_pad;  // padded input channels (mult of 32)
    int             cs;       // stored input channels in Xg (mult of 8)
};

// ---------------------------------------------------------------------------
// Fused grouped-MLP chain + channelwise max-pool over the group.
// One workgroup = one (group, row-tile). Chain stays in LDS ping-pong bufs.
// ---------------------------------------------------------------------------
template <int TILE_M, int MAXW>
__global__ __launch_bounds__(THREADS) void fused_mlp_max(
    MlpDesc d, const _Float16* __restrict__ Xg, int Kpts, int tilesPerGroup,
    float* __restrict__ out) {
    __shared__ _Float16 __attribute__((aligned(16))) buf0[TILE_M * MAXW];
    __shared__ _Float16 __attribute__((aligned(16))) buf1[TILE_M * MAXW];

    const int tid  = threadIdx.x;
    const int lane = tid & 31;
    const int wave = tid >> 5;
    const int g    = blockIdx.x / tilesPerGroup;
    const int t0   = blockIdx.x % tilesPerGroup;
    const int row0 = t0 * TILE_M;
    const int cin0 = d.cin_pad;
    const int cs   = d.cs;

    // Zero the (padded) input tile.
    {
        int4  z    = {0, 0, 0, 0};
        int   nvec = TILE_M * (cin0 >> 3);
        int4* b4   = (int4*)buf0;
        for (int i = tid; i < nvec; i += THREADS) b4[i] = z;
    }
    __syncthreads();
    // Overlay stored channels with async global->LDS DMA (ASYNCcnt path).
    {
        const int       csv = cs >> 3;  // 16B chunks per row
        const _Float16* src = Xg + ((size_t)g * Kpts + row0) * cs;
        for (int i = tid; i < TILE_M * csv; i += THREADS) {
            int             k  = i / csv;
            int             c  = i - k * csv;
            const _Float16* ga = src + (size_t)i * 8;
            unsigned        lo = (unsigned)(unsigned long long)(buf0 + k * cin0 + (c << 3));
            asm volatile("global_load_async_to_lds_b128 %0, %1, off"
                         :: "v"(lo), "v"(ga) : "memory");
        }
        asm volatile("s_wait_asynccnt 0x0" ::: "memory");
    }
    __syncthreads();

    _Float16* X   = buf0;
    _Float16* Y   = buf1;
    int       cin = cin0;
    for (int L = 0; L < d.nlayers; ++L) {
        const int       cout = d.width[L];
        const _Float16* Wl   = d.W[L];
        const float*    bl   = d.bias[L];
        const float*    gl   = d.gamma[L];
        const float*    el   = d.beta[L];
        const int       mtiles = TILE_M / 16;
        const int       ntiles = cout >> 4;
        for (int t = wave; t < mtiles * ntiles; t += NWAVES) {
            int mi = (t % mtiles) * 16;
            int nt = t / mtiles;
            v8f acc = {0.f, 0.f, 0.f, 0.f, 0.f, 0.f, 0.f, 0.f};
            for (int k = 0; k < cin; k += 32) {
                // Fragment-packed weights: tile (kt,nt) = 512 halfs, lane-major.
                const _Float16* wp =
                    Wl + (((size_t)(k >> 5) * ntiles + nt) << 9) + (lane << 4);
                __builtin_prefetch(wp + ((size_t)ntiles << 9), 0, 1);
                v16h b = *(const v16h*)wp;              // 2x global_load_b128
                v16h a = load_a_frag(X, cin, mi, k, lane);
                acc = __builtin_amdgcn_wmma_f32_16x16x32_f16(
                    false, a, false, b, (short)0, acc, false, false);
            }
            // Epilogue: folded batchnorm (scale/shift) + ReLU, write f16.
            int   n  = (nt << 4) + (lane & 15);
            float bb = bl[n], gg = gl[n], ee = el[n];
            int   mr = mi + ((lane >> 4) << 3);
#pragma unroll
            for (int v = 0; v < 8; ++v) {
                float val = fmaxf((acc[v] + bb) * gg + ee, 0.0f);
                Y[(mr + v) * cout + n] = (_Float16)val;
            }
        }
        __syncthreads();
        _Float16* tmp = X; X = Y; Y = tmp;
        cin = cout;
    }

    // Channelwise max over the TILE_M rows; combine across tiles atomically.
    // Post-ReLU values are >= 0 and `out` is pre-zeroed, so int-punned
    // atomicMax is exact.
    const int cf = d.width[d.nlayers - 1];
    for (int c = tid; c < cf; c += THREADS) {
        float m = 0.0f;
        for (int r = 0; r < TILE_M; ++r) m = fmaxf(m, (float)X[r * cf + c]);
        atomicMax((int*)&out[(size_t)g * cf + c], __float_as_int(m));
    }
}

// ---------------------------------------------------------------------------
// Support kernels
// ---------------------------------------------------------------------------
__global__ void zero_f32(float* p, int n) {
    int t = blockIdx.x * blockDim.x + threadIdx.x;
    if (t < n) p[t] = 0.0f;
}

// Pack W f32[cin][cout] -> f16 in WMMA B-fragment order:
// per (kt,nt) tile of 32x16: 32 lanes x 16 contiguous halfs, where lane's
// half j is W[kt*32 + (lane>>4)*8 + j + (j>>3)*8][nt*16 + (lane&15)] (zero K-padding).
__global__ void pack_w_frag(const float* __restrict__ W, _Float16* __restrict__ out,
                            int cin, int cin_pad, int cout) {
    int t = blockIdx.x * blockDim.x + threadIdx.x;
    int total = cin_pad * cout;
    if (t >= total) return;
    int tile = t >> 9;           // 512 halfs per 32x16 tile
    int rem  = t & 511;
    int lane = rem >> 4;
    int j    = rem & 15;
    int ntn  = cout >> 4;
    int kt   = tile / ntn;
    int nt   = tile - kt * ntn;
    int k    = (kt << 5) + ((lane >> 4) << 3) + j + ((j >> 3) << 3);
    int n    = (nt << 4) + (lane & 15);
    out[t]   = (_Float16)((k < cin) ? W[k * cout + n] : 0.0f);
}

// xyz [B][6][N] -> pos [B][N][3], norm [B][N][3]
__global__ void transpose_xyz(const float* __restrict__ xyz, float* __restrict__ pos,
                              float* __restrict__ nrm, int N) {
    int t = blockIdx.x * blockDim.x + threadIdx.x;
    int total = 4 * 6 * N;
    if (t >= total) return;
    int n = t % N;
    int c = (t / N) % 6;
    int b = t / (6 * N);
    float v = xyz[t];
    if (c < 3) pos[((size_t)b * N + n) * 3 + c] = v;
    else       nrm[((size_t)b * N + n) * 3 + (c - 3)] = v;
}

// Farthest-point sampling: one workgroup per batch, serial argmax rounds.
__global__ __launch_bounds__(THREADS) void fps_kernel(const float* __restrict__ pos,
                                                      int N, int npoint, int* __restrict__ out) {
    __shared__ float dist[2048];
    __shared__ float rmax[THREADS];
    __shared__ int   rarg[THREADS];
    __shared__ int   cur;
    const int b = blockIdx.x, tid = threadIdx.x;
    for (int i = tid; i < N; i += THREADS) dist[i] = 1e10f;
    if (tid == 0) cur = 0;
    __syncthreads();
    for (int s = 0; s < npoint; ++s) {
        int far = cur;
        if (tid == 0) out[b * npoint + s] = far;
        const float* cp = pos + ((size_t)b * N + far) * 3;
        float cx = cp[0], cy = cp[1], cz = cp[2];
        float best = -1.0f;
        int   barg = 0;
        for (int i = tid; i < N; i += THREADS) {
            const float* q  = pos + ((size_t)b * N + i) * 3;
            float dx = q[0] - cx, dy = q[1] - cy, dz = q[2] - cz;
            float nd = fminf(dist[i], dx * dx + dy * dy + dz * dz);
            dist[i] = nd;
            if (nd > best) { best = nd; barg = i; }
        }
        rmax[tid] = best;
        rarg[tid] = barg;
        __syncthreads();
        for (int off = THREADS / 2; off > 0; off >>= 1) {
            if (tid < off && rmax[tid + off] > rmax[tid]) {
                rmax[tid] = rmax[tid + off];
                rarg[tid] = rarg[tid + off];
            }
            __syncthreads();
        }
        if (tid == 0) cur = rarg[0];
        __syncthreads();
    }
}

// out[b][s][c] = pos[b][idx[b][s]][c]
__global__ void gather_coords(const float* __restrict__ pos, int npts,
                              const int* __restrict__ idx, int S, float* __restrict__ out) {
    int t = blockIdx.x * blockDim.x + threadIdx.x;
    if (t >= 4 * S * 3) return;
    int c = t % 3;
    int s = (t / 3) % S;
    int b = t / (3 * S);
    out[t] = pos[((size_t)b * npts + idx[b * S + s]) * 3 + c];
}

// First-nsample-in-radius, padded with the first hit (matches reference).
__global__ void ball_query(const float* __restrict__ pos, int npts,
                           const int* __restrict__ centers, int S, float r2,
                           int nsample, int* __restrict__ out) {
    int gidx = blockIdx.x * blockDim.x + threadIdx.x;
    if (gidx >= 4 * S) return;
    int b = gidx / S;
    const float* cp = pos + ((size_t)b * npts + centers[gidx]) * 3;
    float cx = cp[0], cy = cp[1], cz = cp[2];
    int* o = out + (size_t)gidx * nsample;
    int  cnt = 0, first = 0;
    for (int i = 0; i < npts && cnt < nsample; ++i) {
        const float* q  = pos + ((size_t)b * npts + i) * 3;
        float dx = q[0] - cx, dy = q[1] - cy, dz = q[2] - cz;
        if (dx * dx + dy * dy + dz * dz <= r2) {
            if (cnt == 0) first = i;
            o[cnt++] = i;
        }
    }
    for (; cnt < nsample; ++cnt) o[cnt] = first;
}

// Gather grouped [g][k][c] = { centered xyz (3) | feats (cf) | 0-pad } as f16.
__global__ void gather_grouped(const float* __restrict__ pos, int npts,
                               const float* __restrict__ feats, int cf,
                               const int* __restrict__ centers,
                               const int* __restrict__ ballidx, int S, int K,
                               int cs, _Float16* __restrict__ out) {
    int g = blockIdx.x;  // over 4*S
    int b = g / S;
    const float* cp = pos + ((size_t)b * npts + centers[g]) * 3;
    float cx = cp[0], cy = cp[1], cz = cp[2];
    for (int e = threadIdx.x; e < K * cs; e += blockDim.x) {
        int k = e / cs;
        int c = e - k * cs;
        int pi = ballidx[(size_t)g * K + k];
        float v = 0.0f;
        if (c < 3) {
            v = pos[((size_t)b * npts + pi) * 3 + c] - ((c == 0) ? cx : (c == 1) ? cy : cz);
        } else if (c < 3 + cf) {
            v = feats[((size_t)b * npts + pi) * cf + (c - 3)];
        }
        out[(size_t)g * K * cs + e] = (_Float16)v;
    }
}

// group_all gather: [b][k][c] = { xyz (3, uncentered) | feats (cf) | 0-pad }
__global__ void gather_all(const float* __restrict__ pos, const float* __restrict__ feats,
                           int cf, int npts, int cs, _Float16* __restrict__ out) {
    int b = blockIdx.x;
    for (int e = threadIdx.x; e < npts * cs; e += blockDim.x) {
        int k = e / cs;
        int c = e - k * cs;
        float v = 0.0f;
        if (c < 3)            v = pos[((size_t)b * npts + k) * 3 + c];
        else if (c < 3 + cf)  v = feats[((size_t)b * npts + k) * cf + (c - 3)];
        out[(size_t)b * npts * cs + e] = (_Float16)v;
    }
}

// sa4 input: concat([l11x|l11p](64 rows), [l2x|l2p](512 rows)) -> [b][576][264]
__global__ void gather_sa4(const float* __restrict__ l11x, const float* __restrict__ l11p,
                           const float* __restrict__ l2x, const float* __restrict__ l2p,
                           _Float16* __restrict__ out) {
    const int K = 576, cs = 264;
    int b = blockIdx.x;
    for (int e = threadIdx.x; e < K * cs; e += blockDim.x) {
        int k = e / cs;
        int c = e - k * cs;
        float v = 0.0f;
        if (k < 64) {
            if (c < 3)        v = l11x[((size_t)b * 64 + k) * 3 + c];
            else if (c < 259) v = l11p[((size_t)b * 64 + k) * 256 + (c - 3)];
        } else {
            int kk = k - 64;
            if (c < 3)        v = l2x[((size_t)b * 512 + kk) * 3 + c];
            else if (c < 259) v = l2p[((size_t)b * 512 + kk) * 256 + (c - 3)];
        }
        out[(size_t)b * K * cs + e] = (_Float16)v;
    }
}

// FC head (tiny): x=[l3p|l4p] -> fc1(relu) -> fc2(relu) -> fc3 -> log_softmax.
// Also copies l3p (the second tuple output) into d_out+160.
__global__ __launch_bounds__(THREADS) void head_kernel(
    const float* __restrict__ l3p, const float* __restrict__ l4p,
    const float* __restrict__ W1, const float* __restrict__ b1,
    const float* __restrict__ g1, const float* __restrict__ e1,
    const float* __restrict__ W2, const float* __restrict__ b2,
    const float* __restrict__ g2, const float* __restrict__ e2,
    const float* __restrict__ W3, const float* __restrict__ b3,
    float* __restrict__ out) {
    __shared__ float xb[1536];
    __shared__ float h1[512];
    __shared__ float h2[256];
    __shared__ float lg[40];
    const int tid = threadIdx.x;
    for (int b = 0; b < 4; ++b) {
        for (int i = tid; i < 512; i += THREADS)  xb[i]       = l3p[b * 512 + i];
        for (int i = tid; i < 1024; i += THREADS) xb[512 + i] = l4p[b * 1024 + i];
        __syncthreads();
        for (int o = tid; o < 512; o += THREADS) {
            float s = 0.0f;
            for (int i = 0; i < 1536; ++i) s += xb[i] * W1[i * 512 + o];
            h1[o] = fmaxf((s + b1[o]) * g1[o] + e1[o], 0.0f);
        }
        __syncthreads();
        for (int o = tid; o < 256; o += THREADS) {
            float s = 0.0f;
            for (int i = 0; i < 512; ++i) s += h1[i] * W2[i * 256 + o];
            h2[o] = fmaxf((s + b2[o]) * g2[o] + e2[o], 0.0f);
        }
        __syncthreads();
        for (int o = tid; o < 40; o += THREADS) {
            float s = 0.0f;
            for (int i = 0; i < 256; ++i) s += h2[i] * W3[i * 40 + o];
            lg[o] = s + b3[o];
        }
        __syncthreads();
        if (tid == 0) {
            float mx = lg[0];
            for (int i = 1; i < 40; ++i) mx = fmaxf(mx, lg[i]);
            float sum = 0.0f;
            for (int i = 0; i < 40; ++i) sum += __expf(lg[i] - mx);
            float lse = mx + __logf(sum);
            for (int i = 0; i < 40; ++i) out[b * 40 + i] = lg[i] - lse;
        }
        __syncthreads();
    }
    // Second tuple output: transpose(l3p,(0,2,1)) is flat-identical to l3p.
    for (int i = tid; i < 4 * 512; i += THREADS) out[160 + i] = l3p[i];
}

// ---------------------------------------------------------------------------
// Host side
// ---------------------------------------------------------------------------
struct PackInfo { int in_idx; int cin; int cin_pad; int cout; };

extern "C" void kernel_launch(void* const* d_in, const int* in_sizes, int n_in,
                              void* d_out, int out_size, void* d_ws, size_t ws_size,
                              hipStream_t stream) {
    (void)in_sizes; (void)n_in; (void)out_size; (void)ws_size;
    const int B = 4, N = 2048;

    // Input index table: dict order {xyz, sa1, sa1_1, sa2, sa3, sa4, fc1, fc2, fc3},
    // each SA layer is (W, b, g, be).
    const float* xyz = (const float*)d_in[0];
    static const PackInfo packs[16] = {
        // sa1: 6->64->64->128
        {1, 6, 32, 64},    {5, 64, 64, 64},    {9, 64, 64, 128},
        // sa1_1: 131->128->128->256
        {13, 131, 160, 128}, {17, 128, 128, 128}, {21, 128, 128, 256},
        // sa2: 6->64->128->256
        {25, 6, 32, 64},   {29, 64, 64, 128},  {33, 128, 128, 256},
        // sa3: 6->64->128->256->512
        {37, 6, 32, 64},   {41, 64, 64, 128},  {45, 128, 128, 256}, {49, 256, 256, 512},
        // sa4: 259->256->512->1024
        {53, 259, 288, 256}, {57, 256, 256, 512}, {61, 512, 512, 1024},
    };

    // Workspace carve (256B aligned).
    char* wp = (char*)d_ws;
    auto alloc = [&](size_t bytes) {
        void* r = (void*)wp;
        wp += (bytes + 255) & ~(size_t)255;
        return r;
    };
    float* pos    = (float*)alloc((size_t)B * N * 3 * 4);
    float* nrm    = (float*)alloc((size_t)B * N * 3 * 4);
    int*   fps512 = (int*)alloc((size_t)B * 512 * 4);
    int*   fps64  = (int*)alloc((size_t)B * 64 * 4);
    float* l1x    = (float*)alloc((size_t)B * 512 * 3 * 4);   // == l2x
    float* l11x   = (float*)alloc((size_t)B * 64 * 3 * 4);
    int*   ball1  = (int*)alloc((size_t)B * 512 * 256 * 4);
    int*   ball2  = (int*)alloc((size_t)B * 512 * 512 * 4);
    int*   ball11 = (int*)alloc((size_t)B * 64 * 256 * 4);
    _Float16* g1buf  = (_Float16*)alloc((size_t)B * 512 * 256 * 8 * 2);
    _Float16* g2buf  = (_Float16*)alloc((size_t)B * 512 * 512 * 8 * 2);
    _Float16* g11buf = (_Float16*)alloc((size_t)B * 64 * 256 * 136 * 2);
    _Float16* g3buf  = (_Float16*)alloc((size_t)B * 2048 * 8 * 2);
    _Float16* g4buf  = (_Float16*)alloc((size_t)B * 576 * 264 * 2);
    // Pooled outputs: one contiguous zero-initialized region.
    const int poolN = B * 512 * 128 + B * 64 * 256 + B * 512 * 256 + B * 512 + B * 1024;
    float* pooled = (float*)alloc((size_t)poolN * 4);
    float* l1p  = pooled;
    float* l11p = l1p + B * 512 * 128;
    float* l2p  = l11p + B * 64 * 256;
    float* l3p  = l2p + B * 512 * 256;
    float* l4p  = l3p + B * 512;
    // Packed f16 weights (fragment order).
    _Float16* pw[16];
    for (int i = 0; i < 16; ++i)
        pw[i] = (_Float16*)alloc((size_t)packs[i].cin_pad * packs[i].cout * 2);

    // --- 0: pack weights (independent) ---
    for (int i = 0; i < 16; ++i) {
        int total = packs[i].cin_pad * packs[i].cout;
        pack_w_frag<<<(total + THREADS - 1) / THREADS, THREADS, 0, stream>>>(
            (const float*)d_in[packs[i].in_idx], pw[i], packs[i].cin, packs[i].cin_pad,
            packs[i].cout);
    }
    // --- 1: transpose inputs ---
    transpose_xyz<<<(B * 6 * N + THREADS - 1) / THREADS, THREADS, 0, stream>>>(xyz, pos, nrm, N);
    // --- 2: FPS (sa1 and sa2 share _fps(pos,512)) ---
    fps_kernel<<<B, THREADS, 0, stream>>>(pos, N, 512, fps512);
    gather_coords<<<(B * 512 * 3 + THREADS - 1) / THREADS, THREADS, 0, stream>>>(pos, N, fps512, 512, l1x);
    fps_kernel<<<B, THREADS, 0, stream>>>(l1x, 512, 64, fps64);
    gather_coords<<<(B * 64 * 3 + THREADS - 1) / THREADS, THREADS, 0, stream>>>(l1x, 512, fps64, 64, l11x);
    // --- 3: ball queries ---
    ball_query<<<(B * 512 + THREADS - 1) / THREADS, THREADS, 0, stream>>>(
        pos, N, fps512, 512, 0.2f * 0.2f, 256, ball1);
    ball_query<<<(B * 512 + THREADS - 1) / THREADS, THREADS, 0, stream>>>(
        pos, N, fps512, 512, 0.4f * 0.4f, 512, ball2);
    ball_query<<<(B * 64 + THREADS - 1) / THREADS, THREADS, 0, stream>>>(
        l1x, 512, fps64, 64, 0.4f * 0.4f, 256, ball11);
    // --- 4: zero pooled accumulators (required every call) ---
    zero_f32<<<(poolN + THREADS - 1) / THREADS, THREADS, 0, stream>>>(pooled, poolN);
    // --- 5: gathers available now ---
    gather_grouped<<<B * 512, THREADS, 0, stream>>>(pos, N, nrm, 3, fps512, ball1, 512, 256, 8, g1buf);
    gather_grouped<<<B * 512, THREADS, 0, stream>>>(pos, N, nrm, 3, fps512, ball2, 512, 512, 8, g2buf);
    gather_all<<<B, THREADS, 0, stream>>>(pos, nrm, 3, 2048, 8, g3buf);

    // --- 6: MLP chains ---
    auto makeDesc = [&](int lo, int n, int cs) {
        MlpDesc d{};
        for (int i = 0; i < n; ++i) {
            const PackInfo& pi = packs[lo + i];
            d.W[i]     = pw[lo + i];
            d.bias[i]  = (const float*)d_in[pi.in_idx + 1];
            d.gamma[i] = (const float*)d_in[pi.in_idx + 2];
            d.beta[i]  = (const float*)d_in[pi.in_idx + 3];
            d.width[i] = pi.cout;
        }
        d.nlayers = n;
        d.cin_pad = packs[lo].cin_pad;
        d.cs      = cs;
        return d;
    };
    MlpDesc dsa1  = makeDesc(0, 3, 8);
    MlpDesc dsa11 = makeDesc(3, 3, 136);
    MlpDesc dsa2  = makeDesc(6, 3, 8);
    MlpDesc dsa3  = makeDesc(9, 4, 8);
    MlpDesc dsa4  = makeDesc(13, 3, 264);

    fused_mlp_max<64, 128><<<B * 512 * (256 / 64), THREADS, 0, stream>>>(dsa1, g1buf, 256, 256 / 64, l1p);
    gather_grouped<<<B * 64, THREADS, 0, stream>>>(l1x, 512, l1p, 128, fps64, ball11, 64, 256, 136, g11buf);
    fused_mlp_max<64, 256><<<B * 64 * (256 / 64), THREADS, 0, stream>>>(dsa11, g11buf, 256, 256 / 64, l11p);
    fused_mlp_max<64, 256><<<B * 512 * (512 / 64), THREADS, 0, stream>>>(dsa2, g2buf, 512, 512 / 64, l2p);
    fused_mlp_max<32, 512><<<B * (2048 / 32), THREADS, 0, stream>>>(dsa3, g3buf, 2048, 2048 / 32, l3p);
    gather_sa4<<<B, THREADS, 0, stream>>>(l11x, l11p, l1x, l2p, g4buf);
    fused_mlp_max<16, 1024><<<B * (576 / 16), THREADS, 0, stream>>>(dsa4, g4buf, 576, 576 / 16, l4p);

    // --- 7: FC head + log_softmax + second output copy ---
    head_kernel<<<1, THREADS, 0, stream>>>(
        l3p, l4p,
        (const float*)d_in[65], (const float*)d_in[66], (const float*)d_in[67], (const float*)d_in[68],
        (const float*)d_in[69], (const float*)d_in[70], (const float*)d_in[71], (const float*)d_in[72],
        (const float*)d_in[73], (const float*)d_in[74],
        (float*)d_out);
}
